// Harp_81346680586803
// MI455X (gfx1250) — compile-verified, matrix operand order checked by reading, not run on hardware
//
#include <hip/hip_runtime.h>
#include <hip/hip_bf16.h>

// ---------------------------------------------------------------------------
// Problem constants (match reference): B=4, S=2048, D=1024
// ---------------------------------------------------------------------------
#define BD   4
#define SD   2048
#define DD   1024

typedef __attribute__((ext_vector_type(16))) _Float16 v16h;
typedef __attribute__((ext_vector_type(8)))  _Float16 v8h;
typedef __attribute__((ext_vector_type(8)))  float    v8f;

// Load one 16x32 f16 WMMA fragment row for this lane.
// Per ISA 7.12.2 (16-bit A/B 16x32): lanes 0-15 hold row r=lane, K={0..7,16..23};
// lanes 16-31 hold row r=lane-16, K={8..15,24..31}.  Caller passes p already
// offset by (row * ld + khalf*8); we grab halves [0..7] and [16..23].
__device__ __forceinline__ v16h load_frag(const _Float16* __restrict__ p) {
  union { v16h v; v8h h[2]; } u;
  u.h[0] = *(const v8h*)(p);
  u.h[1] = *(const v8h*)(p + 16);
  return u.v;
}

// ---------------------------------------------------------------------------
// Generic batched WMMA GEMM:
//   C[z][M,N] = scale * (A[z][M,K] x Bm[z][N,K]^T) + bias[N]
//   A row-major (lda), Bm row-major along K (ldb); z = blockIdx.z batch with
//   element strides strideA/strideB/strideC (0 => shared across batches).
//   Exactly one of Cf / Ch / Ct is non-null:
//     Cf : f32 row-major   Cf[m*ldc + n]
//     Ch : f16 row-major   Ch[m*ldc + n]
//     Ct : f16 transposed  Ct[n*ldc + m]     (used to build V^T)
//   Block = 256 threads = 8 waves; each wave computes a 32(M) x 64(N) tile
//   (2 M-subtiles x 4 N-subtiles = 8 WMMAs per k-step, A/B frags reused).
//   Block tile = 256(M) x 64(N).  Requires M%256==0, N%64==0, K%32==0.
// ---------------------------------------------------------------------------
__global__ __launch_bounds__(256) void wmma_gemm_f16(
    const _Float16* __restrict__ A, const _Float16* __restrict__ Bm,
    const float* __restrict__ bias, float scale, int K, int lda, int ldb,
    size_t strideA, size_t strideB, size_t strideC,
    float* __restrict__ Cf, _Float16* __restrict__ Ch,
    _Float16* __restrict__ Ct, int ldc)
{
  const int z = blockIdx.z;
  A  += (size_t)z * strideA;
  Bm += (size_t)z * strideB;
  const size_t coff = (size_t)z * strideC;

  const int lane = threadIdx.x & 31;
  const int wave = threadIdx.x >> 5;
  const int row0 = blockIdx.y * 256 + wave * 32;
  const int col0 = blockIdx.x * 64;
  const int r    = lane & 15;
  const int kh   = lane >> 4;           // which K-half group this lane holds

  const _Float16* ap0 = A  + (size_t)(row0 +  0 + r) * lda + kh * 8;
  const _Float16* ap1 = A  + (size_t)(row0 + 16 + r) * lda + kh * 8;
  const _Float16* bp0 = Bm + (size_t)(col0 +  0 + r) * ldb + kh * 8;
  const _Float16* bp1 = Bm + (size_t)(col0 + 16 + r) * ldb + kh * 8;
  const _Float16* bp2 = Bm + (size_t)(col0 + 32 + r) * ldb + kh * 8;
  const _Float16* bp3 = Bm + (size_t)(col0 + 48 + r) * ldb + kh * 8;

  v8f acc00 = {}, acc01 = {}, acc02 = {}, acc03 = {};
  v8f acc10 = {}, acc11 = {}, acc12 = {}, acc13 = {};

  for (int k = 0; k < K; k += 32) {
    v16h a0 = load_frag(ap0 + k);
    v16h a1 = load_frag(ap1 + k);
    v16h b0 = load_frag(bp0 + k);
    v16h b1 = load_frag(bp1 + k);
    v16h b2 = load_frag(bp2 + k);
    v16h b3 = load_frag(bp3 + k);
    acc00 = __builtin_amdgcn_wmma_f32_16x16x32_f16(false, a0, false, b0, (short)0, acc00, false, false);
    acc01 = __builtin_amdgcn_wmma_f32_16x16x32_f16(false, a0, false, b1, (short)0, acc01, false, false);
    acc02 = __builtin_amdgcn_wmma_f32_16x16x32_f16(false, a0, false, b2, (short)0, acc02, false, false);
    acc03 = __builtin_amdgcn_wmma_f32_16x16x32_f16(false, a0, false, b3, (short)0, acc03, false, false);
    acc10 = __builtin_amdgcn_wmma_f32_16x16x32_f16(false, a1, false, b0, (short)0, acc10, false, false);
    acc11 = __builtin_amdgcn_wmma_f32_16x16x32_f16(false, a1, false, b1, (short)0, acc11, false, false);
    acc12 = __builtin_amdgcn_wmma_f32_16x16x32_f16(false, a1, false, b2, (short)0, acc12, false, false);
    acc13 = __builtin_amdgcn_wmma_f32_16x16x32_f16(false, a1, false, b3, (short)0, acc13, false, false);
  }

  // C/D layout (ISA 7.12.2, 32-bit 16x16): lanes 0-15 -> N=lane, M=vgpr;
  // lanes 16-31 -> N=lane-16, M=8+vgpr.
  const int nloc = lane & 15;
  const int mb   = (lane >> 4) * 8;

#define STORE_TILE(ACC, MOFF, TOFF)                                         \
  {                                                                         \
    const int n  = col0 + (TOFF) + nloc;                                    \
    const float bvv = bias ? bias[n] : 0.0f;                                \
    _Pragma("unroll")                                                       \
    for (int i = 0; i < 8; ++i) {                                           \
      const int m = row0 + (MOFF) + mb + i;                                 \
      const float v = ACC[i] * scale + bvv;                                 \
      if (Cf)       Cf[coff + (size_t)m * ldc + n] = v;                     \
      else if (Ch)  Ch[coff + (size_t)m * ldc + n] = (_Float16)v;           \
      else          Ct[coff + (size_t)n * ldc + m] = (_Float16)v;           \
    }                                                                       \
  }
  STORE_TILE(acc00,  0,  0)
  STORE_TILE(acc01,  0, 16)
  STORE_TILE(acc02,  0, 32)
  STORE_TILE(acc03,  0, 48)
  STORE_TILE(acc10, 16,  0)
  STORE_TILE(acc11, 16, 16)
  STORE_TILE(acc12, 16, 32)
  STORE_TILE(acc13, 16, 48)
#undef STORE_TILE
}

// ---------------------------------------------------------------------------
// f32 -> f16 conversion
// ---------------------------------------------------------------------------
__global__ void f32_to_f16(const float* __restrict__ in, _Float16* __restrict__ out, int n) {
  int i = blockIdx.x * 256 + threadIdx.x;
  if (i < n) out[i] = (_Float16)in[i];
}

// ---------------------------------------------------------------------------
// Row softmax over S=2048 (f32 scores -> f16 attn).  One block (256) per row.
// ---------------------------------------------------------------------------
__global__ __launch_bounds__(256) void softmax_rows(
    const float* __restrict__ scores, _Float16* __restrict__ attn)
{
  __shared__ float red[256];
  const size_t base = (size_t)blockIdx.x * SD;
  float v[8];
  float mx = -3.4e38f;
#pragma unroll
  for (int i = 0; i < 8; ++i) {
    v[i] = scores[base + threadIdx.x + i * 256];
    mx = fmaxf(mx, v[i]);
  }
  red[threadIdx.x] = mx; __syncthreads();
  for (int s = 128; s > 0; s >>= 1) {
    if (threadIdx.x < s) red[threadIdx.x] = fmaxf(red[threadIdx.x], red[threadIdx.x + s]);
    __syncthreads();
  }
  mx = red[0]; __syncthreads();
  float sum = 0.0f;
#pragma unroll
  for (int i = 0; i < 8; ++i) { v[i] = __expf(v[i] - mx); sum += v[i]; }
  red[threadIdx.x] = sum; __syncthreads();
  for (int s = 128; s > 0; s >>= 1) {
    if (threadIdx.x < s) red[threadIdx.x] += red[threadIdx.x + s];
    __syncthreads();
  }
  const float inv = 1.0f / red[0];
#pragma unroll
  for (int i = 0; i < 8; ++i)
    attn[base + threadIdx.x + i * 256] = (_Float16)(v[i] * inv);
}

// ---------------------------------------------------------------------------
// Gate: p[b] = sigmoid( mean_s( h[b,s,:] . Wg ) + bg ).  One block per batch.
// ---------------------------------------------------------------------------
__global__ __launch_bounds__(256) void gate_kernel(
    const _Float16* __restrict__ hb, const float* __restrict__ Wg,
    const float* __restrict__ bg, float* __restrict__ pg)
{
  __shared__ float red[256];
  const int b = blockIdx.x;
  float acc = 0.0f;
  for (int s = threadIdx.x; s < SD; s += 256) {
    const _Float16* hp = hb + ((size_t)b * SD + s) * DD;
    float dot = 0.0f;
    for (int d = 0; d < DD; d += 8) {
      v8h hv = *(const v8h*)(hp + d);
#pragma unroll
      for (int j = 0; j < 8; ++j) dot += (float)hv[j] * Wg[d + j];
    }
    acc += dot;
  }
  red[threadIdx.x] = acc; __syncthreads();
  for (int s = 128; s > 0; s >>= 1) {
    if (threadIdx.x < s) red[threadIdx.x] += red[threadIdx.x + s];
    __syncthreads();
  }
  if (threadIdx.x == 0)
    pg[b] = 1.0f / (1.0f + __expf(-(red[0] / (float)SD + bg[0])));
}

// ---------------------------------------------------------------------------
// Fused reparameterize + residual + LayerNorm.  One block (256) per row of D=1024.
//   t = x + relu(mu + eps * exp(0.5*logvar));  out = LN(t)*gamma + beta
// ---------------------------------------------------------------------------
__global__ __launch_bounds__(256) void fuse_reparam_ln(
    const float* __restrict__ x, const float* __restrict__ eps,
    const float* __restrict__ mu, const float* __restrict__ lv,
    const float* __restrict__ gamma, const float* __restrict__ beta,
    float* __restrict__ out)
{
  __shared__ float red[256];
  const size_t base = (size_t)blockIdx.x * DD;
  float t[4];
  float sum = 0.0f;
#pragma unroll
  for (int i = 0; i < 4; ++i) {
    const int j = threadIdx.x + i * 256;
    const float sd = __expf(0.5f * lv[base + j]);
    const float z  = fmaxf(mu[base + j] + eps[base + j] * sd, 0.0f);
    t[i] = x[base + j] + z;
    sum += t[i];
  }
  red[threadIdx.x] = sum; __syncthreads();
  for (int s = 128; s > 0; s >>= 1) {
    if (threadIdx.x < s) red[threadIdx.x] += red[threadIdx.x + s];
    __syncthreads();
  }
  const float mean = red[0] * (1.0f / (float)DD);
  __syncthreads();
  float sq = 0.0f;
#pragma unroll
  for (int i = 0; i < 4; ++i) { const float d = t[i] - mean; sq += d * d; }
  red[threadIdx.x] = sq; __syncthreads();
  for (int s = 128; s > 0; s >>= 1) {
    if (threadIdx.x < s) red[threadIdx.x] += red[threadIdx.x + s];
    __syncthreads();
  }
  const float inv = rsqrtf(red[0] * (1.0f / (float)DD) + 1e-5f);
#pragma unroll
  for (int i = 0; i < 4; ++i) {
    const int j = threadIdx.x + i * 256;
    out[base + j] = (t[i] - mean) * inv * gamma[j] + beta[j];
  }
}

// ---------------------------------------------------------------------------
// Host launcher
// ---------------------------------------------------------------------------
extern "C" void kernel_launch(void* const* d_in, const int* in_sizes, int n_in,
                              void* d_out, int out_size, void* d_ws, size_t ws_size,
                              hipStream_t stream) {
  (void)in_sizes; (void)n_in; (void)out_size; (void)ws_size;

  const float* x    = (const float*)d_in[0];
  const float* eps  = (const float*)d_in[1];
  const float* Wq   = (const float*)d_in[2];
  const float* bq   = (const float*)d_in[3];
  const float* Wk   = (const float*)d_in[4];
  const float* bk   = (const float*)d_in[5];
  const float* Wv   = (const float*)d_in[6];
  const float* bv   = (const float*)d_in[7];
  const float* Wmu  = (const float*)d_in[8];
  const float* bmu  = (const float*)d_in[9];
  const float* Wlv  = (const float*)d_in[10];
  const float* blv  = (const float*)d_in[11];
  const float* Wg   = (const float*)d_in[12];
  const float* bg   = (const float*)d_in[13];
  const float* gam  = (const float*)d_in[14];
  const float* bet  = (const float*)d_in[15];

  const size_t BSD = (size_t)BD * SD * DD;   // 8388608
  const size_t DD2 = (size_t)DD * DD;        // 1048576
  const size_t BSS = (size_t)BD * SD * SD;   // 16777216

  float* outp  = (float*)d_out;          // [0, BSD)       : out
  float* mu_o  = outp + BSD;             // [BSD, 2BSD)    : mu
  float* lv_o  = outp + 2 * BSD;         // [2BSD, 3BSD)   : logvar
  float* pg_o  = outp + 3 * BSD;         // [3BSD, 3BSD+B) : p_gate

  // Workspace carve-up (f16 intermediates + f32 scores)
  char* ws = (char*)d_ws;
  size_t off = 0;
  _Float16* xb   = (_Float16*)(ws + off); off += BSD * 2;
  _Float16* Wqb  = (_Float16*)(ws + off); off += DD2 * 2;
  _Float16* Wkb  = (_Float16*)(ws + off); off += DD2 * 2;
  _Float16* Wvb  = (_Float16*)(ws + off); off += DD2 * 2;
  _Float16* Wmub = (_Float16*)(ws + off); off += DD2 * 2;
  _Float16* Wlvb = (_Float16*)(ws + off); off += DD2 * 2;
  _Float16* Qb   = (_Float16*)(ws + off); off += BSD * 2;
  _Float16* Kb   = (_Float16*)(ws + off); off += BSD * 2;
  _Float16* Vt   = (_Float16*)(ws + off); off += BSD * 2;   // V^T per batch [D][S]
  _Float16* hb   = (_Float16*)(ws + off); off += BSD * 2;
  _Float16* attn = (_Float16*)(ws + off); off += BSS * 2;
  float*    sco  = (float*)   (ws + off); off += BSS * 4;

  const float inv_scale = 1.0f / (32.0f * 1.5f);  // 1/(sqrt(1024)*1.5)

  // 1) convert to f16
  f32_to_f16<<<(int)((BSD + 255) / 256), 256, 0, stream>>>(x,   xb,   (int)BSD);
  f32_to_f16<<<(int)((DD2 + 255) / 256), 256, 0, stream>>>(Wq,  Wqb,  (int)DD2);
  f32_to_f16<<<(int)((DD2 + 255) / 256), 256, 0, stream>>>(Wk,  Wkb,  (int)DD2);
  f32_to_f16<<<(int)((DD2 + 255) / 256), 256, 0, stream>>>(Wv,  Wvb,  (int)DD2);
  f32_to_f16<<<(int)((DD2 + 255) / 256), 256, 0, stream>>>(Wmu, Wmub, (int)DD2);
  f32_to_f16<<<(int)((DD2 + 255) / 256), 256, 0, stream>>>(Wlv, Wlvb, (int)DD2);

  const dim3 blk(256);

  // 2) Q = x Wq^T + bq ; K = x Wk^T + bk  (M=8192, N=1024, K=1024) -> f16
  {
    dim3 grid(DD / 64, (BD * SD) / 256, 1);
    wmma_gemm_f16<<<grid, blk, 0, stream>>>(xb, Wqb, bq, 1.0f, DD, DD, DD,
                                            0, 0, 0, nullptr, Qb, nullptr, DD);
    wmma_gemm_f16<<<grid, blk, 0, stream>>>(xb, Wkb, bk, 1.0f, DD, DD, DD,
                                            0, 0, 0, nullptr, Kb, nullptr, DD);
  }
  // 3) V stored transposed per batch: Vt[b][d][s]  (batched over z)
  {
    dim3 grid(DD / 64, SD / 256, BD);
    wmma_gemm_f16<<<grid, blk, 0, stream>>>(xb, Wvb, bv, 1.0f, DD, DD, DD,
                                            (size_t)SD * DD, 0, (size_t)DD * SD,
                                            nullptr, nullptr, Vt, SD);
  }
  // 4) scores[b] = Q[b] K[b]^T * inv_scale  (M=2048, N=2048, K=1024) -> f32
  {
    dim3 grid(SD / 64, SD / 256, BD);
    wmma_gemm_f16<<<grid, blk, 0, stream>>>(Qb, Kb, nullptr, inv_scale, DD, DD, DD,
                                            (size_t)SD * DD, (size_t)SD * DD, (size_t)SD * SD,
                                            sco, nullptr, nullptr, SD);
  }
  // 5) softmax rows -> f16 attn
  softmax_rows<<<BD * SD, blk, 0, stream>>>(sco, attn);

  // 6) h[b] = attn[b] V[b]  (M=2048, N=1024, K=2048) -> f16, contract over t via Vt rows
  {
    dim3 grid(DD / 64, SD / 256, BD);
    wmma_gemm_f16<<<grid, blk, 0, stream>>>(attn, Vt, nullptr, 1.0f, SD, SD, SD,
                                            (size_t)SD * SD, (size_t)DD * SD, (size_t)SD * DD,
                                            nullptr, hb, nullptr, DD);
  }
  // 7) mu / logvar (M=8192, N=1024, K=1024) -> f32 straight into d_out
  {
    dim3 grid(DD / 64, (BD * SD) / 256, 1);
    wmma_gemm_f16<<<grid, blk, 0, stream>>>(hb, Wmub, bmu, 1.0f, DD, DD, DD,
                                            0, 0, 0, mu_o, nullptr, nullptr, DD);
    wmma_gemm_f16<<<grid, blk, 0, stream>>>(hb, Wlvb, blv, 1.0f, DD, DD, DD,
                                            0, 0, 0, lv_o, nullptr, nullptr, DD);
  }
  // 8) gate
  gate_kernel<<<BD, blk, 0, stream>>>(hb, Wg, bg, pg_o);

  // 9) fused reparam + residual + LayerNorm -> out
  fuse_reparam_ln<<<BD * SD, blk, 0, stream>>>(x, eps, mu_o, lv_o, gam, bet, outp);
}
